// WassersteinLoss_27968827031705
// MI455X (gfx1250) — matrix operand order repaired and checked by reading, not compiled, not run on hardware
//
#include <hip/hip_runtime.h>
#include <math.h>

// ============================================================================
// Sinkhorn entropic OT (reg=1, 100 iters) for 4096x1024 fp32 clouds, gfx1250.
//
// Cost GEMM: error-compensated bf16 split (x = hi + lo) using
// V_WMMA_F32_16X16X32_BF16:  dot(x,y) ~= hi*hi + hi*lo + lo*hi  (3 WMMAs per
// 32-deep K chunk = 96 WMMAs/tile vs 256 for fp32 WMMA, 8x FLOP/instruction).
// Dropped lo*lo term is ~2^-32 relative; C-matrix error ~2^-17 relative ->
// ~2e-5 relative error in the final scalar. Memory-neutral vs fp32.
//
// C matrix (64 MB) kept in d_ws -> L2-resident (192 MB) across the 200
// logsumexp passes of the Sinkhorn loop (L2-bandwidth bound, not HBM).
// ============================================================================

typedef __attribute__((ext_vector_type(16))) __bf16 v16bf;
typedef __attribute__((ext_vector_type(8)))  __bf16 v8bf;
typedef __attribute__((ext_vector_type(8)))  float  v8f;

#define D_FEAT 1024
#define REG_V  1.0f
#define N_ITERS 100

// ---------------------------------------------------------------- row norms
__global__ void row_norm_kernel(const float* __restrict__ X,
                                float* __restrict__ out, int D) {
  __shared__ float red[256];
  int row = blockIdx.x;
  const float* p = X + (size_t)row * D;
  float s = 0.f;
  for (int j = threadIdx.x; j < D; j += 256) { float v = p[j]; s += v * v; }
  red[threadIdx.x] = s;
  for (int w = 128; w > 0; w >>= 1) {
    __syncthreads();
    if (threadIdx.x < w) red[threadIdx.x] += red[threadIdx.x + w];
  }
  if (threadIdx.x == 0) out[row] = red[0];
}

// ------------------------------------------------------------------- zeroing
__global__ void zero_kernel(float* __restrict__ p, int n) {
  int i = blockIdx.x * 256 + threadIdx.x;
  if (i < n) p[i] = 0.f;
}

// --------------------------------------------- fp32 -> (bf16 hi, bf16 lo)
__global__ void split_bf16_kernel(const float* __restrict__ X,
                                  __bf16* __restrict__ hi,
                                  __bf16* __restrict__ lo, int n) {
  int i = blockIdx.x * 256 + threadIdx.x;
  if (i < n) {
    float v = X[i];
    __bf16 h = (__bf16)v;
    hi[i] = h;
    lo[i] = (__bf16)(v - (float)h);
  }
}

// ----------------------------------------- cost GEMM (bf16-split WMMA)
// C[i][j] = sx[i] + sy[j] - 2 * dot(X[i,:], Y[j,:])
// One wave per 16x16 tile; 8 waves (256 thr) per block -> 16 rows x 128 cols.
// A-frag (bf16 16x32, v16bf): lanes 0-15 row M=l hold K = k+[0..7] (elems 0-7)
//   and K = k+16+[0..7] (elems 8-15); lanes 16-31 hold K = k+8+[0..7] and
//   K = k+24+[0..7]  -> two contiguous b128 loads per lane.
// B-frag (bf16 32x16): same K layout as A-frag (lanes 0-15 col N=l hold
//   K = k+[0..7], k+16+[0..7]; lanes 16-31 K = k+8+[0..7], k+24+[0..7]) -> 2x b128.
// D (16x16 f32): VGPR v -> row (v + 8*(lane>=16)), col = lane&15.
__global__ void cost_gemm_bf16_kernel(const __bf16* __restrict__ Xhi,
                                      const __bf16* __restrict__ Xlo,
                                      const __bf16* __restrict__ Yhi,
                                      const __bf16* __restrict__ Ylo,
                                      const float* __restrict__ sx,
                                      const float* __restrict__ sy,
                                      float* __restrict__ C, int M) {
  const int wave = threadIdx.x >> 5;
  const int lane = threadIdx.x & 31;
  const int gsel = lane >> 4;
  const int l    = lane & 15;
  const int ti = blockIdx.y;              // 16-row tile
  const int tj = blockIdx.x * 8 + wave;   // 16-col tile

  const size_t arow = (size_t)(ti * 16 + l) * D_FEAT;   // A row (X)
  const size_t brow = (size_t)(tj * 16 + l) * D_FEAT;   // B col (Y row)
  const int aoff = 8 * gsel;     // A K-subchunk for this half-wave
  const int boff = 8 * gsel;     // B K-subchunk for this half-wave

  union Frag { v16bf v; v8bf h[2]; };

  v8f acc = {};
#pragma unroll 2
  for (int k = 0; k < D_FEAT; k += 32) {
    Frag ah, al, bh, bl;
    ah.h[0] = *(const v8bf*)(Xhi + arow + k + aoff);        // b128
    ah.h[1] = *(const v8bf*)(Xhi + arow + k + aoff + 16);   // b128
    al.h[0] = *(const v8bf*)(Xlo + arow + k + aoff);
    al.h[1] = *(const v8bf*)(Xlo + arow + k + aoff + 16);
    bh.h[0] = *(const v8bf*)(Yhi + brow + k + boff);
    bh.h[1] = *(const v8bf*)(Yhi + brow + k + boff + 16);
    bl.h[0] = *(const v8bf*)(Ylo + brow + k + boff);
    bl.h[1] = *(const v8bf*)(Ylo + brow + k + boff + 16);
    // dot += lo_a*hi_b + hi_a*lo_b + hi_a*hi_b   (accumulator chained via C)
    acc = __builtin_amdgcn_wmma_f32_16x16x32_bf16(
        false, al.v, false, bh.v, (short)0, acc, false, false);
    acc = __builtin_amdgcn_wmma_f32_16x16x32_bf16(
        false, ah.v, false, bl.v, (short)0, acc, false, false);
    acc = __builtin_amdgcn_wmma_f32_16x16x32_bf16(
        false, ah.v, false, bh.v, (short)0, acc, false, false);
  }

  const int ccol  = tj * 16 + l;
  const int rbase = ti * 16 + gsel * 8;
  const float syc = sy[ccol];
#pragma unroll
  for (int v = 0; v < 8; ++v) {
    const int i = rbase + v;
    C[(size_t)i * M + ccol] = sx[i] + syc - 2.0f * acc[v];
  }
}

// ------------------------------------------------------------ LSE utilities
__device__ __forceinline__ void lse_merge(float& m1, float& s1,
                                          float m2, float s2) {
  if (m2 > m1) { s1 = s1 * __expf(m1 - m2) + s2; m1 = m2; }
  else if (m2 != -INFINITY) { s1 += s2 * __expf(m2 - m1); }
}

__device__ __forceinline__ void lse_push(float& mx, float& s, float v) {
  if (v > mx) { s = s * __expf(mx - v) + 1.f; mx = v; }
  else        { s += __expf(v - mx); }
}

// ------------------------------------------- row pass: f_i from g (per row)
// f_i = -reg * ( LSE_j[(g_j - C_ij)/reg] - log m )
__global__ void row_lse_kernel(const float* __restrict__ C,
                               const float* __restrict__ g,
                               float* __restrict__ f,
                               int M, float invReg, float logM) {
  __shared__ float smx[256], ssm[256];
  const int i = blockIdx.x;
  const float* row = C + (size_t)i * M;
  float mx = -INFINITY, s = 0.f;
  for (int j = threadIdx.x; j < M; j += 256)
    lse_push(mx, s, (g[j] - row[j]) * invReg);
  smx[threadIdx.x] = mx; ssm[threadIdx.x] = s;
  for (int w = 128; w > 0; w >>= 1) {
    __syncthreads();
    if (threadIdx.x < w) {
      float m1 = smx[threadIdx.x], s1 = ssm[threadIdx.x];
      lse_merge(m1, s1, smx[threadIdx.x + w], ssm[threadIdx.x + w]);
      smx[threadIdx.x] = m1; ssm[threadIdx.x] = s1;
    }
  }
  if (threadIdx.x == 0)
    f[i] = -REG_V * (smx[0] + __logf(ssm[0]) - logM);
}

// ---------------------------- column pass, stage 1: partial LSE per (chunk,j)
// Block covers 64 consecutive columns x (N/gridDim.y) rows; warps read 32
// consecutive columns per row -> coalesced 128B segments out of L2.
__global__ void col_lse_partial_kernel(const float* __restrict__ C,
                                       const float* __restrict__ f,
                                       float2* __restrict__ part,
                                       int N, int M, float invReg) {
  __shared__ float smx[256], ssm[256];
  const int c = threadIdx.x & 63;
  const int r = threadIdx.x >> 6;                 // 0..3
  const int j = blockIdx.x * 64 + c;
  const int chunkRows = N / gridDim.y;
  const int i0 = blockIdx.y * chunkRows;
  float mx = -INFINITY, s = 0.f;
  for (int i = i0 + r; i < i0 + chunkRows; i += 4)
    lse_push(mx, s, (f[i] - C[(size_t)i * M + j]) * invReg);
  smx[threadIdx.x] = mx; ssm[threadIdx.x] = s;
  __syncthreads();
  if (r == 0) {
    float m1 = mx, s1 = s;
#pragma unroll
    for (int rr = 1; rr < 4; ++rr)
      lse_merge(m1, s1, smx[c + 64 * rr], ssm[c + 64 * rr]);
    part[(size_t)blockIdx.y * M + j] = make_float2(m1, s1);
  }
}

// ---------------------------- column pass, stage 2: combine chunks -> g_j
__global__ void col_combine_kernel(const float2* __restrict__ part,
                                   float* __restrict__ g,
                                   int M, int nchunk, float logN) {
  const int j = blockIdx.x * 256 + threadIdx.x;
  if (j >= M) return;
  float mx = -INFINITY, s = 0.f;
  for (int c = 0; c < nchunk; ++c) {
    float2 p = part[(size_t)c * M + j];
    lse_merge(mx, s, p.x, p.y);
  }
  g[j] = -REG_V * (mx + __logf(s) - logN);
}

// ------------------- final: rowval[i] = sum_j P_ij * (f_i + g_j)
// logP_ij = (f_i + g_j - C_ij)/reg + log a + log b ;  lab = -(log n + log m)
__global__ void final_row_kernel(const float* __restrict__ C,
                                 const float* __restrict__ f,
                                 const float* __restrict__ g,
                                 float* __restrict__ rowval,
                                 int M, float invReg, float lab) {
  __shared__ float red[256];
  const int i = blockIdx.x;
  const float fi = f[i];
  const float* row = C + (size_t)i * M;
  float acc = 0.f;
  for (int j = threadIdx.x; j < M; j += 256) {
    const float fg = fi + g[j];
    acc += __expf((fg - row[j]) * invReg + lab) * fg;
  }
  red[threadIdx.x] = acc;
  for (int w = 128; w > 0; w >>= 1) {
    __syncthreads();
    if (threadIdx.x < w) red[threadIdx.x] += red[threadIdx.x + w];
  }
  if (threadIdx.x == 0) rowval[i] = red[0];
}

__global__ void final_reduce_kernel(const float* __restrict__ rowval,
                                    float* __restrict__ out,
                                    int N, float scale) {
  __shared__ float red[256];
  float s = 0.f;
  for (int i = threadIdx.x; i < N; i += 256) s += rowval[i];
  red[threadIdx.x] = s;
  for (int w = 128; w > 0; w >>= 1) {
    __syncthreads();
    if (threadIdx.x < w) red[threadIdx.x] += red[threadIdx.x + w];
  }
  if (threadIdx.x == 0) out[0] = red[0] * scale;
}

// ============================================================================
extern "C" void kernel_launch(void* const* d_in, const int* in_sizes, int n_in,
                              void* d_out, int out_size, void* d_ws,
                              size_t ws_size, hipStream_t stream) {
  (void)n_in; (void)out_size; (void)ws_size;
  const float* X = (const float*)d_in[0];
  const float* Y = (const float*)d_in[1];
  float* out = (float*)d_out;

  const int N = in_sizes[0] / D_FEAT;   // 4096
  const int M = in_sizes[1] / D_FEAT;   // 4096
  const int NCHUNK = 8;
  const int nX = in_sizes[0], nY = in_sizes[1];

  // ---- workspace carve-out (C matrix first: 64 MB, L2-resident) ----
  char* ws = (char*)d_ws;
  size_t off = 0;
  auto carve = [&](size_t bytes) -> void* {
    void* p = ws + off;
    off = (off + bytes + 255) & ~(size_t)255;
    return p;
  };
  float*  Cm     = (float*)carve((size_t)N * M * sizeof(float));
  __bf16* Xhi    = (__bf16*)carve((size_t)nX * sizeof(__bf16));
  __bf16* Xlo    = (__bf16*)carve((size_t)nX * sizeof(__bf16));
  __bf16* Yhi    = (__bf16*)carve((size_t)nY * sizeof(__bf16));
  __bf16* Ylo    = (__bf16*)carve((size_t)nY * sizeof(__bf16));
  float*  sx     = (float*)carve((size_t)N * sizeof(float));
  float*  sy     = (float*)carve((size_t)M * sizeof(float));
  float*  f      = (float*)carve((size_t)N * sizeof(float));
  float*  g      = (float*)carve((size_t)M * sizeof(float));
  float2* part   = (float2*)carve((size_t)NCHUNK * M * sizeof(float2));
  float*  rowval = (float*)carve((size_t)N * sizeof(float));

  const float invReg = 1.0f / REG_V;
  const float logN = logf((float)N);
  const float logM = logf((float)M);
  const float lab = -(logN + logM);

  // 1) squared norms (exact fp32) + bf16 hi/lo decomposition
  row_norm_kernel<<<N, 256, 0, stream>>>(X, sx, D_FEAT);
  row_norm_kernel<<<M, 256, 0, stream>>>(Y, sy, D_FEAT);
  split_bf16_kernel<<<(nX + 255) / 256, 256, 0, stream>>>(X, Xhi, Xlo, nX);
  split_bf16_kernel<<<(nY + 255) / 256, 256, 0, stream>>>(Y, Yhi, Ylo, nY);

  // 2) cost matrix via bf16-split WMMA (3x 16x16x32 per K-chunk)
  cost_gemm_bf16_kernel<<<dim3(M / 128, N / 16), 256, 0, stream>>>(
      Xhi, Xlo, Yhi, Ylo, sx, sy, Cm, M);

  // 3) Sinkhorn loop (log domain): f <- from g, g <- from new f
  zero_kernel<<<(M + 255) / 256, 256, 0, stream>>>(g, M);
  for (int it = 0; it < N_ITERS; ++it) {
    row_lse_kernel<<<N, 256, 0, stream>>>(Cm, g, f, M, invReg, logM);
    col_lse_partial_kernel<<<dim3(M / 64, NCHUNK), 256, 0, stream>>>(
        Cm, f, part, N, M, invReg);
    col_combine_kernel<<<(M + 255) / 256, 256, 0, stream>>>(part, g, M,
                                                            NCHUNK, logN);
  }

  // 4) value = sum_ij P_ij (f_i + g_j), divided by feature dim
  final_row_kernel<<<N, 256, 0, stream>>>(Cm, f, g, rowval, M, invReg, lab);
  final_reduce_kernel<<<1, 256, 0, stream>>>(rowval, out, N,
                                             1.0f / (float)D_FEAT);
}
